// BaseDeepBSDE_57896159150567
// MI455X (gfx1250) — compile-verified, matrix operand order checked by reading, not run on hardware
//
#include <hip/hip_runtime.h>

// DeepBSDE forward: B=16384 paths, N=100 steps, two MLPs (2->64->64->{3,1}) per step.
// Each wave owns 16 samples persistently across all steps (paths are independent),
// uses v_wmma_f32_16x16x32_f16 for the 64x64 layers, and atomically accumulates loss.

#define B_PATHS   16384
#define N_STEPS   100
#define DT_C      0.01f
#define SQRT_DT_C 0.1f
#define SIGMA0_C  0.5f

typedef __attribute__((ext_vector_type(16))) _Float16 v16h;
typedef __attribute__((ext_vector_type(8)))  float    v8f;

union Frag { v16h h; unsigned u[8]; };

// DS ops within a wave are in-order; the fence stops compiler reordering and
// drains DScnt before dependent cross-lane LDS reads.
#define LDS_FENCE() asm volatile("s_wait_dscnt 0x0" ::: "memory")

__device__ __forceinline__ unsigned pack2f16(float a, float b) {
  union { _Float16 h[2]; unsigned u; } t;
  t.h[0] = (_Float16)a; t.h[1] = (_Float16)b;
  return t.u;
}

__device__ __forceinline__ v8f wmma16(const Frag& a, const Frag& b, v8f c) {
  // (neg_a, A, neg_b, B, c_mod, C, reuse_a, reuse_b)
  return __builtin_amdgcn_wmma_f32_16x16x32_f16(false, a.h, false, b.h,
                                                (short)0, c, false, false);
}

// One MLP evaluation for this wave's 16 samples.
// hb: per-wave LDS staging tile (16x64 f16, reused h1 -> h2).
// Layer1 is a rank-1 update written straight into A-fragment-feedable layout.
// Returns layer-3 C fragment (16x16, only first 3 / 1 columns meaningful).
__device__ __forceinline__ v8f mlp_eval(
    _Float16* hb, const float* yarr, float t,
    float w1t0, float w1t1, float w1y0, float w1y1, float b10, float b11,
    const float (&b2v)[4], const Frag (&B2)[2][4], const Frag (&B3)[2],
    int lane, int hf, int m)
{
  const int   k0 = lane * 2;                 // this lane owns columns k0, k0+1
  const float c0 = t * w1t0 + b10;
  const float c1 = t * w1t1 + b11;
#pragma unroll
  for (int r = 0; r < 16; r++) {
    float h0 = fmaxf(fmaf(yarr[r], w1y0, c0), 0.f);
    float h1 = fmaxf(fmaf(yarr[r], w1y1, c1), 0.f);
    *reinterpret_cast<unsigned*>(hb + r * 64 + k0) = pack2f16(h0, h1);
  }
  LDS_FENCE();

  // Load A fragments (16x32 f16 each; lane l holds row m=l&15,
  // K in {0..7,16..23} for lanes 0-15 and {8..15,24..31} for lanes 16-31).
  Frag a0, a1;
#pragma unroll
  for (int p = 0; p < 8; p++) {
    const int kb = (p & 3) * 2 + 8 * hf + 16 * (p >> 2);
    a0.u[p] = *reinterpret_cast<const unsigned*>(hb + m * 64 + kb);
    a1.u[p] = *reinterpret_cast<const unsigned*>(hb + m * 64 + 32 + kb);
  }
  LDS_FENCE();

  // Layer 2: 4 N-tiles x 2 K-chunks = 8 WMMAs, bias+relu, restage as f16.
#pragma unroll
  for (int nt = 0; nt < 4; nt++) {
    v8f c = {0.f,0.f,0.f,0.f,0.f,0.f,0.f,0.f};
    c = wmma16(a0, B2[0][nt], c);
    c = wmma16(a1, B2[1][nt], c);
#pragma unroll
    for (int v = 0; v < 8; v++) {
      float h = fmaxf(c[v] + b2v[nt], 0.f);       // C layout: M=v+8*hf, N=nt*16+m
      hb[(v + 8 * hf) * 64 + nt * 16 + m] = (_Float16)h;
    }
  }
  LDS_FENCE();

  // Layer 3: reload A fragments from h2, 2 WMMAs against padded W3.
  Frag e0, e1;
#pragma unroll
  for (int p = 0; p < 8; p++) {
    const int kb = (p & 3) * 2 + 8 * hf + 16 * (p >> 2);
    e0.u[p] = *reinterpret_cast<const unsigned*>(hb + m * 64 + kb);
    e1.u[p] = *reinterpret_cast<const unsigned*>(hb + m * 64 + 32 + kb);
  }
  v8f c3 = {0.f,0.f,0.f,0.f,0.f,0.f,0.f,0.f};
  c3 = wmma16(e0, B3[0], c3);
  c3 = wmma16(e1, B3[1], c3);
  return c3;
}

__global__ __launch_bounds__(256) void deepbsde_kernel(
    const float* __restrict__ y0,  const float* __restrict__ Y0,
    const float* __restrict__ zW1, const float* __restrict__ zb1,
    const float* __restrict__ zW2, const float* __restrict__ zb2,
    const float* __restrict__ zW3, const float* __restrict__ zb3,
    const float* __restrict__ qW1, const float* __restrict__ qb1,
    const float* __restrict__ qW2, const float* __restrict__ qb2,
    const float* __restrict__ qW3, const float* __restrict__ qb3,
    const float* __restrict__ dW,  const float* __restrict__ dZ,
    float* __restrict__ out)
{
  __shared__ _Float16 hbuf[8][16 * 64];   // per-wave staging tile
  __shared__ float    outz[8][16 * 3];
  __shared__ float    outq[8][16];
  __shared__ float    yw  [8][16];
  __shared__ float    red [8][32];

  const int tid  = threadIdx.x;
  const int wv   = tid >> 5;      // wave in block (wave32)
  const int lane = tid & 31;
  const int hf   = lane >> 4;     // half-wave
  const int m    = lane & 15;     // sample row within wave tile

  _Float16* hb = &hbuf[wv][0];

  // ---- One-time weight preload into registers ----
  const int   k0    = lane * 2;
  const float zw1t0 = zW1[k0],      zw1t1 = zW1[k0 + 1];
  const float zw1y0 = zW1[64 + k0], zw1y1 = zW1[64 + k0 + 1];
  const float zb10  = zb1[k0],      zb11  = zb1[k0 + 1];
  const float qw1t0 = qW1[k0],      qw1t1 = qW1[k0 + 1];
  const float qw1y0 = qW1[64 + k0], qw1y1 = qW1[64 + k0 + 1];
  const float qb10  = qb1[k0],      qb11  = qb1[k0 + 1];

  float zb2v[4], qb2v[4];
#pragma unroll
  for (int nt = 0; nt < 4; nt++) { zb2v[nt] = zb2[nt * 16 + m]; qb2v[nt] = qb2[nt * 16 + m]; }
  const float zb30 = zb3[0], zb31 = zb3[1], zb32 = zb3[2], qb30 = qb3[0];

  // B fragments (32x16 f16): lane l holds column N=l&15; K mirrors the A map.
  Frag zB2[2][4], qB2[2][4], zB3[2], qB3[2];
#pragma unroll
  for (int kc = 0; kc < 2; kc++)
#pragma unroll
    for (int nt = 0; nt < 4; nt++) {
      const int N = nt * 16 + m;
#pragma unroll
      for (int p = 0; p < 8; p++) {
        const int K = kc * 32 + (p & 3) * 2 + 8 * hf + 16 * (p >> 2);
        zB2[kc][nt].u[p] = pack2f16(zW2[K * 64 + N], zW2[(K + 1) * 64 + N]);
        qB2[kc][nt].u[p] = pack2f16(qW2[K * 64 + N], qW2[(K + 1) * 64 + N]);
      }
    }
#pragma unroll
  for (int kc = 0; kc < 2; kc++) {
#pragma unroll
    for (int p = 0; p < 8; p++) {
      const int K = kc * 32 + (p & 3) * 2 + 8 * hf + 16 * (p >> 2);
      const float za  = (m < 3) ? zW3[K * 3 + m]       : 0.f;
      const float zbv = (m < 3) ? zW3[(K + 1) * 3 + m] : 0.f;
      zB3[kc].u[p] = pack2f16(za, zbv);
      const float qa  = (m == 0) ? qW3[K]     : 0.f;
      const float qb  = (m == 0) ? qW3[K + 1] : 0.f;
      qB3[kc].u[p] = pack2f16(qa, qb);
    }
  }

  // ---- Per-path state ----
  const int sample = blockIdx.x * 128 + wv * 16 + m;
  float y    = y0[0];
  float Yv   = Y0[0];
  float loss = 0.f;
  yw[wv][m] = y;                  // both half-waves write identical values
  LDS_FENCE();

  for (int i = 0; i < N_STEPS; i++) {
    const float t = (float)i * DT_C;

    if (i + 1 < N_STEPS) {        // prefetch next step's Brownian increments
      const size_t offn = ((size_t)(i + 1) * B_PATHS + sample) * 3;
      __builtin_prefetch(dW + offn, 0, 1);
      __builtin_prefetch(dZ + offn, 0, 1);
    }

    float yarr[16];
#pragma unroll
    for (int r = 0; r < 16; r++) yarr[r] = yw[wv][r];

    const v8f cz = mlp_eval(hb, yarr, t, zw1t0, zw1t1, zw1y0, zw1y1, zb10, zb11,
                            zb2v, zB2, zB3, lane, hf, m);
    if (m < 3) {
#pragma unroll
      for (int v = 0; v < 8; v++) outz[wv][(v + 8 * hf) * 3 + m] = cz[v];
    }
    const v8f cq = mlp_eval(hb, yarr, t, qw1t0, qw1t1, qw1y0, qw1y1, qb10, qb11,
                            qb2v, qB2, qB3, lane, hf, m);
    if (m == 0) {
#pragma unroll
      for (int v = 0; v < 8; v++) outq[wv][v + 8 * hf] = cq[v];
    }
    LDS_FENCE();

    if (hf == 0) {                // lanes 0-15 own the per-sample state update
      const float z0 = outz[wv][m * 3 + 0] + zb30;
      const float z1 = outz[wv][m * 3 + 1] + zb31;
      const float z2 = outz[wv][m * 3 + 2] + zb32;
      const float q  = outq[wv][m]         + qb30;
      const size_t off = ((size_t)i * B_PATHS + sample) * 3;
      const float dw0 = dW[off + 0] * SQRT_DT_C, dw1 = dW[off + 1] * SQRT_DT_C,
                  dw2 = dW[off + 2] * SQRT_DT_C;
      const float dz0 = dZ[off + 0] * SQRT_DT_C, dz1 = dZ[off + 1] * SQRT_DT_C,
                  dz2 = dZ[off + 2] * SQRT_DT_C;
      const float f       = 0.5f * q * q;
      const float sum_zdw = z0 * dw0 + z1 * dw1 + z2 * dw2;
      const float sum_zdz = z0 * dz0 + z1 * dz1 + z2 * dz2;
      const float resid   = sum_zdw - sum_zdz;     // Y - f*dt terms cancel
      loss += resid * resid;
      y  = y + q * DT_C + SIGMA0_C * (dw0 + dw1 + dw2);
      Yv = Yv - f * DT_C + sum_zdw;
      yw[wv][m] = y;
    }
    LDS_FENCE();
  }

  if (hf == 0) {
    const float term = Yv - y * y;
    loss += term * term;
  }
  red[wv][lane] = (hf == 0) ? loss : 0.f;
  LDS_FENCE();
  if (lane == 0) {
    float s = 0.f;
#pragma unroll
    for (int l = 0; l < 32; l++) s += red[wv][l];
    atomicAdd(out, s * (1.0f / (float)B_PATHS));
  }
}

__global__ void deepbsde_zero_out(float* o) { o[0] = 0.f; }

extern "C" void kernel_launch(void* const* d_in, const int* in_sizes, int n_in,
                              void* d_out, int out_size, void* d_ws, size_t ws_size,
                              hipStream_t stream) {
  (void)in_sizes; (void)n_in; (void)out_size; (void)d_ws; (void)ws_size;
  const float* y0  = (const float*)d_in[0];
  const float* Y0  = (const float*)d_in[1];
  const float* zW1 = (const float*)d_in[2];
  const float* zb1 = (const float*)d_in[3];
  const float* zW2 = (const float*)d_in[4];
  const float* zb2 = (const float*)d_in[5];
  const float* zW3 = (const float*)d_in[6];
  const float* zb3 = (const float*)d_in[7];
  const float* qW1 = (const float*)d_in[8];
  const float* qb1 = (const float*)d_in[9];
  const float* qW2 = (const float*)d_in[10];
  const float* qb2 = (const float*)d_in[11];
  const float* qW3 = (const float*)d_in[12];
  const float* qb3 = (const float*)d_in[13];
  const float* dW  = (const float*)d_in[14];
  const float* dZ  = (const float*)d_in[15];
  float* out = (float*)d_out;

  deepbsde_zero_out<<<1, 1, 0, stream>>>(out);
  // 16384 paths / (16 per wave * 8 waves per block) = 128 blocks
  deepbsde_kernel<<<dim3(128), dim3(256), 0, stream>>>(
      y0, Y0, zW1, zb1, zW2, zb2, zW3, zb3,
      qW1, qb1, qW2, qb2, qW3, qb3, dW, dZ, out);
}
